// Encoder_86071144612137
// MI455X (gfx1250) — matrix operand order repaired and checked by reading, not compiled, not run on hardware
//
#include <hip/hip_runtime.h>
#include <hip/hip_bf16.h>
#include <math.h>

// ---------------- CDNA5 WMMA plumbing (wave32, gfx1250) ----------------
typedef __attribute__((ext_vector_type(16))) __bf16 bf16x16;
typedef __attribute__((ext_vector_type(8)))  float  f32x8;

#define XLS 136  // LDS row stride in bf16 elems (272B = 17*16B, bank-spread, 16B aligned)
#define VTS 24   // k_attn V-tile row stride (48B = 3*16B)

union Half8 { uint4 u; __bf16 h[8]; };
union Half2 { unsigned int u; __bf16 h[2]; };

static __device__ __forceinline__ bf16x16 make_frag(const __bf16* lo, const __bf16* hi) {
  Half8 a, b;
  a.u = *reinterpret_cast<const uint4*>(lo);
  b.u = *reinterpret_cast<const uint4*>(hi);
  bf16x16 f;
#pragma unroll
  for (int j = 0; j < 8; ++j) { f[j] = a.h[j]; f[j + 8] = b.h[j]; }
  return f;
}

// A-matrix 16x32 bf16 fragment from row-major wT[m*128+k] (k contiguous).
// Lanes 0-15 hold K = {0..7, 16..23}, lanes 16-31 hold K = {8..15, 24..31}.
static __device__ __forceinline__ bf16x16 load_afrag(const __bf16* wT, int m, int kstep, int lane) {
  int kb = (lane >= 16) ? 8 : 0;
  const __bf16* p = wT + m * 128 + kstep * 32 + kb;
  return make_frag(p, p + 16);
}

// B-matrix 32x16 bf16 fragment from LDS laid out [n][k] (k contiguous, stride XLS).
// Lanes 0-15 hold K = {0..7, 16..23}, lanes 16-31 hold K = {8..15, 24..31}, n = lane&15.
static __device__ __forceinline__ bf16x16 load_bfrag_lds(const __bf16* Xl, int n, int kstep, int lane) {
  int kb = (lane >= 16) ? 8 : 0;
  const __bf16* p = Xl + n * XLS + kstep * 32 + kb;
  return make_frag(p, p + 16);
}

static __device__ __forceinline__ f32x8 wmma_bf16(bf16x16 a, bf16x16 b, f32x8 c) {
  return __builtin_amdgcn_wmma_f32_16x16x32_bf16(false, a, false, b, (short)0, c, false, false);
}

static __device__ __forceinline__ f32x8 zero8() {
  f32x8 z;
#pragma unroll
  for (int r = 0; r < 8; ++r) z[r] = 0.0f;
  return z;
}

static __device__ __forceinline__ float gelu_f(float x) {
  float x3 = x * x * x;
  return 0.5f * x * (1.0f + tanhf(0.7978845608028654f * (x + 0.044715f * x3)));
}

// Swin region id on the rolled grid (N_SUB=16, WINDOW=4, SHIFT=2).
static __device__ __forceinline__ int swin_id(int win, int t) {
  int gy = ((win >> 2) << 2) + (t >> 2);
  int gx = ((win & 3) << 2) + (t & 3);
  int ry = gy < 12 ? 0 : (gy < 14 ? 1 : 2);
  int rx = gx < 12 ? 0 : (gx < 14 ? 1 : 2);
  return ry * 3 + rx;
}

// Shared GEMM tile-pass: one wave computes rows [m0,m0+16) x 128 cols of one chunk,
// y = wT @ X + bias -> bf16 out.  A-frags/bias loaded just-in-time (keeps VGPRs low).
static __device__ __forceinline__ void gemm_pass_bf16(
    const __bf16* Xl, const __bf16* wT, const float* bias, __bf16* out,
    int m0, int lane, int hi8, int nl, int chunk) {
  bf16x16 a[4];
#pragma unroll
  for (int ks = 0; ks < 4; ++ks) a[ks] = load_afrag(wT, m0 + nl, ks, lane);
  float bb[8];
#pragma unroll
  for (int r = 0; r < 8; ++r) bb[r] = bias[m0 + r + hi8];
  for (int nt = 0; nt < 8; ++nt) {
    bf16x16 bf[4];
#pragma unroll
    for (int ks = 0; ks < 4; ++ks) bf[ks] = load_bfrag_lds(Xl, nt * 16 + nl, ks, lane);
    f32x8 acc;
#pragma unroll
    for (int r = 0; r < 8; ++r) acc[r] = bb[r];
#pragma unroll
    for (int ks = 0; ks < 4; ++ks) acc = wmma_bf16(a[ks], bf[ks], acc);
    int n = chunk * 128 + nt * 16 + nl;
#pragma unroll
    for (int r = 0; r < 8; ++r) out[(size_t)(m0 + r + hi8) * 256 + n] = (__bf16)acc[r];
  }
}

// ---------------- Kernel 0: weights f32 [c_in][c_out] -> bf16 wT[m][k] ----------------
__global__ void k_wprep(const float* w0, const float* w1, const float* w2,
                        const float* w3, const float* w4, const float* w5,
                        __bf16* wt) {
  const float* tbl[6] = {w0, w1, w2, w3, w4, w5};
  const float* w = tbl[blockIdx.x];
  __bf16* o = wt + (size_t)blockIdx.x * 16384;
#pragma unroll 4
  for (int it = 0; it < 64; ++it) {
    int idx = it * 256 + threadIdx.x;
    int m = idx >> 7, kk = idx & 127;
    o[idx] = (__bf16)w[kk * 128 + m];  // wT[m][k] = w[k][m]
  }
}

// ---------------- Kernel 1/5: InstanceNorm (+ optional roll+window permute) ----------------
__global__ void k_inorm(const float* __restrict__ src, __bf16* __restrict__ dst, int permute) {
  int bid  = blockIdx.x;         // (b*256 + s_dst)*16 + cgrp
  int cgrp = bid & 15;
  int bs   = bid >> 4;
  int b    = bs >> 8;
  int sd   = bs & 255;
  int wave = threadIdx.x >> 5;
  int lane = threadIdx.x & 31;
  int c    = cgrp * 8 + wave;

  int ssrc = sd;
  if (permute) {  // dest is windowed order on the rolled grid; source is original grid
    int win = sd >> 4, tok = sd & 15;
    int gy = ((win >> 2) << 2) + (tok >> 2);
    int gx = ((win & 3) << 2) + (tok & 3);
    ssrc = (((gy + 2) & 15) << 4) | ((gx + 2) & 15);  // roll(-2): src = g + 2 (mod 16)
  }

  const float* row = src + (((size_t)b * 256 + ssrc) * 128 + c) * 256;
  float x[8];
  float s0 = 0.f, s1 = 0.f;
#pragma unroll
  for (int i = 0; i < 8; ++i) {
    x[i] = row[lane + i * 32];
    s0 += x[i];
    s1 += x[i] * x[i];
  }
#pragma unroll
  for (int m = 1; m < 32; m <<= 1) {
    s0 += __shfl_xor(s0, m, 32);
    s1 += __shfl_xor(s1, m, 32);
  }
  float mean = s0 * (1.0f / 256.0f);
  float var  = s1 * (1.0f / 256.0f) - mean * mean;
  float rs   = rsqrtf(var + 1e-5f);

  __bf16* orow = dst + (((size_t)b * 256 + sd) * 128 + c) * 256;
#pragma unroll
  for (int i = 0; i < 8; ++i)
    orow[lane + i * 32] = (__bf16)((x[i] - mean) * rs);
}

// ---------------- Kernel 2: fused QKV projection (WMMA) ----------------
__global__ void k_qkv(const __bf16* __restrict__ xn,
                      const __bf16* __restrict__ wtq, const __bf16* __restrict__ wtk,
                      const __bf16* __restrict__ wtv,
                      const float* __restrict__ bq, const float* __restrict__ bk,
                      const float* __restrict__ bv,
                      __bf16* __restrict__ qo, __bf16* __restrict__ ko, __bf16* __restrict__ vo) {
  __shared__ __bf16 Xl[128 * XLS];
  int bs   = blockIdx.x;
  int wid  = threadIdx.x >> 5;
  int lane = threadIdx.x & 31;
  int m0   = wid * 16;
  int hi8  = (lane >= 16) ? 8 : 0;
  int nl   = lane & 15;

  const __bf16* X = xn + (size_t)bs * 32768;
  __bf16* qb = qo + (size_t)bs * 32768;
  __bf16* kb = ko + (size_t)bs * 32768;
  __bf16* vb = vo + (size_t)bs * 32768;

  for (int chunk = 0; chunk < 2; ++chunk) {
    __syncthreads();
    if (chunk == 0) __builtin_prefetch(X + 128, 0, 1);
    // stage + transpose, b32 (2x bf16) per thread: Xl[n][c] = X[c][chunk*128+n]
#pragma unroll 4
    for (int it = 0; it < 32; ++it) {
      int idx = it * 256 + threadIdx.x;       // 8192 uint loads
      int cc = idx >> 6, np = (idx & 63) * 2;
      Half2 h;
      h.u = *reinterpret_cast<const unsigned int*>(X + cc * 256 + chunk * 128 + np);
      Xl[np * XLS + cc]       = h.h[0];
      Xl[(np + 1) * XLS + cc] = h.h[1];
    }
    __syncthreads();
    gemm_pass_bf16(Xl, wtq, bq, qb, m0, lane, hi8, nl, chunk);
    gemm_pass_bf16(Xl, wtk, bk, kb, m0, lane, hi8, nl, chunk);
    gemm_pass_bf16(Xl, wtv, bv, vb, m0, lane, hi8, nl, chunk);
  }
}

// ---------------- Kernel 3: scores + mask + softmax + attn@V per (b, window, head) ----------------
__global__ void k_attn(const __bf16* __restrict__ qm, const __bf16* __restrict__ km,
                       const __bf16* __restrict__ vm, __bf16* __restrict__ om) {
  __shared__ float attnLds[256];        // attn[t][s]
  __shared__ __bf16 Vt[256 * VTS];      // staged V chunk, transposed: Vt[n][tok]
  int bid  = blockIdx.x;                // b*64 + win*4 + head
  int head = bid & 3;
  int win  = (bid >> 2) & 15;
  int b    = bid >> 6;
  int lane = threadIdx.x & 31;
  int hi   = lane >> 4;
  int nl   = lane & 15;
  int kbA  = hi ? 8 : 0;
  int kbB  = hi ? 16 : 0;

  const size_t rowstride = 32768;  // 128*256 elems per token
  size_t base = (((size_t)b * 256 + win * 16) * 128 + head * 32) * 256;

  // ---- scores[t][s] = sum_{k=0..8191} q[t][k]*k[s][k], 256 K-steps of 32 ----
  const __bf16* qrow = qm + base + (size_t)nl * rowstride;  // A: row = token = lane&15
  const __bf16* krow = km + base + (size_t)nl * rowstride;  // B: col = token = lane&15
  f32x8 acc = zero8();
  for (int ks = 0; ks < 256; ++ks) {
    const __bf16* pa = qrow + ks * 32 + kbA;
    const __bf16* pb = krow + ks * 32 + kbA;
    acc = wmma_bf16(make_frag(pa, pa + 16), make_frag(pb, pb + 16), acc);
  }

  // ---- mask + softmax over s (16 lanes within each half-wave) ----
  const float scale = 1.0f / (5.656854249492381f * 256.0f);  // 1/(sqrt(32)*H*W)
  int id_s = swin_id(win, nl);
#pragma unroll
  for (int r = 0; r < 8; ++r) {
    int t = r + hi * 8;
    float x = acc[r] * scale + ((swin_id(win, t) == id_s) ? 0.0f : -1e9f);
    float mx = x;
#pragma unroll
    for (int m = 1; m < 16; m <<= 1) mx = fmaxf(mx, __shfl_xor(mx, m, 32));
    float e = __expf(x - mx);
    float sum = e;
#pragma unroll
    for (int m = 1; m < 16; m <<= 1) sum += __shfl_xor(sum, m, 32);
    attnLds[t * 16 + nl] = e / sum;
  }
  __syncthreads();

  // A fragment: attn zero-padded from K=16 to K=32 (lanes 0-15: K = j, lanes 16-31 zero)
  bf16x16 afr;
#pragma unroll
  for (int j = 0; j < 8; ++j) {
    afr[j]     = hi ? (__bf16)0.0f : (__bf16)attnLds[nl * 16 + j];
    afr[j + 8] = hi ? (__bf16)0.0f : (__bf16)attnLds[nl * 16 + 8 + j];
  }

  // ---- o[t][n] = sum_s attn[t][s] * v[s][n]; V staged through LDS, 256 cols/chunk ----
  for (int c0 = 0; c0 < 8192; c0 += 256) {
    __syncthreads();
    // coalesced stage: each lane pulls 8 contiguous bf16 (16B) per token row
#pragma unroll
    for (int j = 0; j < 16; ++j) {
      Half8 h8;
      h8.u = *reinterpret_cast<const uint4*>(vm + base + (size_t)j * rowstride + c0 + lane * 8);
#pragma unroll
      for (int e = 0; e < 8; ++e) Vt[(lane * 8 + e) * VTS + j] = h8.h[e];
    }
    __syncthreads();
    for (int nt = 0; nt < 16; ++nt) {
      int nloc = nt * 16 + nl;
      bf16x16 bfr;
      if (hi == 0) {  // lanes 0-15 carry K=j (tokens); lanes 16-31 K>=16 -> zero
        const __bf16* p = Vt + nloc * VTS;
        bfr = make_frag(p, p + 8);
      } else {
#pragma unroll
        for (int j = 0; j < 16; ++j) bfr[j] = (__bf16)0.0f;
      }
      f32x8 ao = wmma_bf16(afr, bfr, zero8());
      int n = c0 + nloc;
#pragma unroll
      for (int r = 0; r < 8; ++r)
        om[base + (size_t)(r + hi * 8) * rowstride + n] = (__bf16)ao[r];
    }
  }
}

// ---------------- Kernel 4: O-projection + un-window/un-roll + residual (f32 out) ----------------
__global__ void k_oproj(const __bf16* __restrict__ ob, const __bf16* __restrict__ wto,
                        const float* __restrict__ bo, const float* __restrict__ vres,
                        float* __restrict__ aout) {
  __shared__ __bf16 Xl[128 * XLS];
  int bs   = blockIdx.x;
  int b    = bs >> 8;
  int sd   = bs & 255;
  int wid  = threadIdx.x >> 5;
  int lane = threadIdx.x & 31;
  int m0   = wid * 16;
  int hi8  = (lane >= 16) ? 8 : 0;
  int nl   = lane & 15;

  // windowed-rolled s' -> original grid position (unwindow then roll(+2))
  int win = sd >> 4, tok = sd & 15;
  int gy = ((win >> 2) << 2) + (tok >> 2);
  int gx = ((win & 3) << 2) + (tok & 3);
  int sout = (((gy + 2) & 15) << 4) | ((gx + 2) & 15);

  const __bf16* X = ob + (size_t)bs * 32768;
  size_t obase = ((size_t)b * 256 + sout) * 32768;

  for (int chunk = 0; chunk < 2; ++chunk) {
    __syncthreads();
    if (chunk == 0) __builtin_prefetch(X + 128, 0, 1);
#pragma unroll 4
    for (int it = 0; it < 32; ++it) {
      int idx = it * 256 + threadIdx.x;
      int cc = idx >> 6, np = (idx & 63) * 2;
      Half2 h;
      h.u = *reinterpret_cast<const unsigned int*>(X + cc * 256 + chunk * 128 + np);
      Xl[np * XLS + cc]       = h.h[0];
      Xl[(np + 1) * XLS + cc] = h.h[1];
    }
    __syncthreads();
    bf16x16 a[4];
#pragma unroll
    for (int ks = 0; ks < 4; ++ks) a[ks] = load_afrag(wto, m0 + nl, ks, lane);
    float bb[8];
#pragma unroll
    for (int r = 0; r < 8; ++r) bb[r] = bo[m0 + r + hi8];
    for (int nt = 0; nt < 8; ++nt) {
      bf16x16 bf[4];
#pragma unroll
      for (int ks = 0; ks < 4; ++ks) bf[ks] = load_bfrag_lds(Xl, nt * 16 + nl, ks, lane);
      f32x8 acc;
#pragma unroll
      for (int r = 0; r < 8; ++r) acc[r] = bb[r];
#pragma unroll
      for (int ks = 0; ks < 4; ++ks) acc = wmma_bf16(a[ks], bf[ks], acc);
      int n = chunk * 128 + nt * 16 + nl;
#pragma unroll
      for (int r = 0; r < 8; ++r) {
        size_t oi = obase + (size_t)(m0 + r + hi8) * 256 + n;
        aout[oi] = acc[r] + vres[oi];
      }
    }
  }
}

// ---------------- Kernel 6: fused MLP (gemm1 -> gelu -> gemm2 + residual) ----------------
__global__ void k_mlp(const __bf16* __restrict__ h1,
                      const __bf16* __restrict__ w1t, const __bf16* __restrict__ w2t,
                      const float* __restrict__ b1, const float* __restrict__ b2,
                      const float* __restrict__ resid, float* __restrict__ out) {
  __shared__ __bf16 Xl[64 * XLS];
  __shared__ __bf16 Tl[64 * XLS];
  int bs   = blockIdx.x;
  int wid  = threadIdx.x >> 5;
  int lane = threadIdx.x & 31;
  int m0   = wid * 16;
  int hi8  = (lane >= 16) ? 8 : 0;
  int nl   = lane & 15;

  const __bf16* X = h1 + (size_t)bs * 32768;
  size_t obase = (size_t)bs * 32768;

  for (int chunk = 0; chunk < 4; ++chunk) {
    __syncthreads();
#pragma unroll 4
    for (int it = 0; it < 16; ++it) {
      int idx = it * 256 + threadIdx.x;       // 4096 uint loads for 128x64
      int cc = idx >> 5, np = (idx & 31) * 2;
      Half2 h;
      h.u = *reinterpret_cast<const unsigned int*>(X + cc * 256 + chunk * 64 + np);
      Xl[np * XLS + cc]       = h.h[0];
      Xl[(np + 1) * XLS + cc] = h.h[1];
    }
    __syncthreads();
    // ---- gemm1 + gelu -> Tl[n][m] (A-frags loaded JIT, only one set live) ----
    {
      bf16x16 a[4];
#pragma unroll
      for (int ks = 0; ks < 4; ++ks) a[ks] = load_afrag(w1t, m0 + nl, ks, lane);
      float bb[8];
#pragma unroll
      for (int r = 0; r < 8; ++r) bb[r] = b1[m0 + r + hi8];
      for (int nt = 0; nt < 4; ++nt) {
        bf16x16 bf[4];
#pragma unroll
        for (int ks = 0; ks < 4; ++ks) bf[ks] = load_bfrag_lds(Xl, nt * 16 + nl, ks, lane);
        f32x8 acc;
#pragma unroll
        for (int r = 0; r < 8; ++r) acc[r] = bb[r];
#pragma unroll
        for (int ks = 0; ks < 4; ++ks) acc = wmma_bf16(a[ks], bf[ks], acc);
#pragma unroll
        for (int r = 0; r < 8; ++r)
          Tl[(nt * 16 + nl) * XLS + (m0 + r + hi8)] = (__bf16)gelu_f(acc[r]);
      }
    }
    __syncthreads();
    // ---- gemm2 + residual -> out ----
    {
      bf16x16 a[4];
#pragma unroll
      for (int ks = 0; ks < 4; ++ks) a[ks] = load_afrag(w2t, m0 + nl, ks, lane);
      float bb[8];
#pragma unroll
      for (int r = 0; r < 8; ++r) bb[r] = b2[m0 + r + hi8];
      for (int nt = 0; nt < 4; ++nt) {
        bf16x16 bf[4];
#pragma unroll
        for (int ks = 0; ks < 4; ++ks) bf[ks] = load_bfrag_lds(Tl, nt * 16 + nl, ks, lane);
        f32x8 acc;
#pragma unroll
        for (int r = 0; r < 8; ++r) acc[r] = bb[r];
#pragma unroll
        for (int ks = 0; ks < 4; ++ks) acc = wmma_bf16(a[ks], bf[ks], acc);
        int n = chunk * 64 + nt * 16 + nl;
#pragma unroll
        for (int r = 0; r < 8; ++r) {
          size_t oi = obase + (size_t)(m0 + r + hi8) * 256 + n;
          out[oi] = acc[r] + resid[oi];
        }
      }
    }
  }
}

// ---------------- Host launcher ----------------
extern "C" void kernel_launch(void* const* d_in, const int* in_sizes, int n_in,
                              void* d_out, int out_size, void* d_ws, size_t ws_size,
                              hipStream_t stream) {
  (void)in_sizes; (void)n_in; (void)out_size; (void)ws_size;
  const float* v   = (const float*)d_in[0];
  const float* w_q = (const float*)d_in[1];
  const float* b_q = (const float*)d_in[2];
  const float* w_k = (const float*)d_in[3];
  const float* b_k = (const float*)d_in[4];
  const float* w_v = (const float*)d_in[5];
  const float* b_v = (const float*)d_in[6];
  const float* w_o = (const float*)d_in[7];
  const float* b_o = (const float*)d_in[8];
  const float* w1  = (const float*)d_in[9];
  const float* b1  = (const float*)d_in[10];
  const float* w2  = (const float*)d_in[11];
  const float* b2  = (const float*)d_in[12];

  const size_t MB128 = 134217728ull;  // 8*256*128*256 * 2B
  char* ws = (char*)d_ws;
  // lifetime-overlapped workspace plan (peak 512MB + 192KB):
  __bf16* xn   = (__bf16*)(ws);                 // [0,1) normed+permuted input (bf16)
  __bf16* qb   = (__bf16*)(ws + 1 * MB128);     // [1,2)
  __bf16* kb   = (__bf16*)(ws + 2 * MB128);     // [2,3)
  __bf16* vb   = (__bf16*)(ws + 3 * MB128);     // [3,4)
  __bf16* ob   = xn;                            // attn output reuses xn region
  float*  aout = (float*)(ws + 1 * MB128);      // [1,3) f32, reuses q/k after attn
  __bf16* h1   = (__bf16*)(ws + 3 * MB128);     // [3,4), reuses v after attn
  __bf16* wt   = (__bf16*)(ws + 4 * MB128);     // 6 x 32KB bf16 transposed weights

  k_wprep<<<6, 256, 0, stream>>>(w_q, w_k, w_v, w_o, w1, w2, wt);
  k_inorm<<<32768, 256, 0, stream>>>(v, xn, 1);
  k_qkv<<<2048, 256, 0, stream>>>(xn, wt, wt + 16384, wt + 32768, b_q, b_k, b_v, qb, kb, vb);
  k_attn<<<512, 32, 0, stream>>>(qb, kb, vb, ob);
  k_oproj<<<2048, 256, 0, stream>>>(ob, wt + 49152, b_o, v, aout);
  k_inorm<<<32768, 256, 0, stream>>>(aout, h1, 0);
  k_mlp<<<2048, 256, 0, stream>>>(h1, wt + 65536, wt + 81920, b1, b2, aout, (float*)d_out);
}
